// KANLinear_11158325035415
// MI455X (gfx1250) — matrix operand order repaired and checked by reading, not compiled, not run on hardware
//
#include <hip/hip_runtime.h>

typedef __attribute__((ext_vector_type(16))) __bf16 v16bf;
typedef __attribute__((ext_vector_type(8)))  __bf16 v8bf;
typedef __attribute__((ext_vector_type(8)))  float  v8f;

#define IN_FEAT   1024
#define OUT_FEAT  1024
#define NROWS     8192
#define KDIM      9216      // IN_FEAT * (1 + 8 spline bases)
#define KSTEPS    288       // KDIM / 32
#define LDS_PITCH 40        // bf16 elements per 32-elem tile row (80 B, padded)

#define BM 256              // block tile M
#define BN 128              // block tile N
#define AOFF 0
#define BOFF (BM * LDS_PITCH)                    // element offset of B tile
#define BUFSZ ((BM + BN) * LDS_PITCH)            // elements per double-buffer half

__device__ __forceinline__ __bf16 tobf(float f) { return (__bf16)f; }

// ---------------------------------------------------------------------------
// A[n][0:1024] = silu(x); A[n][1024+8i+k] = cubic B-spline bases
// ---------------------------------------------------------------------------
__global__ __launch_bounds__(256) void kan_prep_a(const float* __restrict__ x,
                                                  const float* __restrict__ grid,
                                                  __bf16* __restrict__ A) {
  const int i = ((blockIdx.x & 3) << 8) + threadIdx.x;
  const int n = blockIdx.x >> 2;
  const float xv = x[(size_t)n * IN_FEAT + i];
  __bf16* row = A + (size_t)n * KDIM;
  row[i] = tobf(xv / (1.0f + __expf(-xv)));

  float g[12];
#pragma unroll
  for (int j = 0; j < 12; ++j) g[j] = grid[i * 12 + j];
  float b[11];
#pragma unroll
  for (int j = 0; j < 11; ++j)
    b[j] = (xv >= g[j] && xv < g[j + 1]) ? 1.0f : 0.0f;
#pragma unroll
  for (int k = 1; k <= 3; ++k) {
#pragma unroll
    for (int j = 0; j + k < 11; ++j) {
      float l = (xv - g[j])         / (g[j + k]     - g[j]);
      float r = (g[j + k + 1] - xv) / (g[j + k + 1] - g[j + 1]);
      b[j] = l * b[j] + r * b[j + 1];
    }
  }
  v8bf pk;
#pragma unroll
  for (int k = 0; k < 8; ++k) pk[k] = tobf(b[k]);
  *reinterpret_cast<v8bf*>(row + IN_FEAT + (size_t)i * 8) = pk;
}

// ---------------------------------------------------------------------------
// B[o][0:1024] = base_weight; B[o][1024+8i+k] = spline_weight * scaler
// ---------------------------------------------------------------------------
__global__ __launch_bounds__(256) void kan_prep_b(const float* __restrict__ bw,
                                                  const float* __restrict__ sw,
                                                  const float* __restrict__ sc,
                                                  __bf16* __restrict__ B) {
  const int i = ((blockIdx.x & 3) << 8) + threadIdx.x;
  const int o = blockIdx.x >> 2;
  const size_t oi = (size_t)o * IN_FEAT + i;
  __bf16* row = B + (size_t)o * KDIM;
  row[i] = tobf(bw[oi]);
  const float s = sc[oi];
  const float* w = sw + oi * 8;
  v8bf pk;
#pragma unroll
  for (int k = 0; k < 8; ++k) pk[k] = tobf(w[k] * s);
  *reinterpret_cast<v8bf*>(row + IN_FEAT + (size_t)i * 8) = pk;
}

// ---------------------------------------------------------------------------
// C[n][o] = sum_k A[n][k]*B[o][k].  Block 256x128, 8 waves of 64x64
// (4x4 WMMA tiles, 128 acc VGPRs).  LDS double buffered, K-step 32.
// ---------------------------------------------------------------------------
__global__ __launch_bounds__(256) void kan_gemm(const __bf16* __restrict__ A,
                                                const __bf16* __restrict__ B,
                                                float* __restrict__ C) {
  __shared__ __align__(16) __bf16 sm[2 * BUFSZ];   // 60 KB

  const int tid   = threadIdx.x;
  const int lane  = tid & 31;
  const int wid   = tid >> 5;
  const int wm    = wid >> 1;          // 0..3  (M quadrant)
  const int wn    = wid & 1;           // 0..1  (N half)
  const int bm    = blockIdx.y;        // 0..31
  const int bn    = blockIdx.x;        // 0..7
  const int lhalf = lane >> 4;         // 0/1
  const int lrow  = lane & 15;

  // ---- global staging: per K-step each thread loads 64B of A, 32B of B ----
  const __bf16* gA = A + (size_t)(bm * BM + tid) * KDIM;          // whole row slab
  const int brow = tid >> 1;
  const int bcol = (tid & 1) << 4;                                 // 0 or 16 elems
  const __bf16* gB = B + (size_t)(bn * BN + brow) * KDIM + bcol;

  const int sAo = AOFF + tid * LDS_PITCH;            // + {0,8,16,24}
  const int sBo = BOFF + brow * LDS_PITCH + bcol;    // + {0,8}

  v8f acc[4][4];
#pragma unroll
  for (int m = 0; m < 4; ++m)
#pragma unroll
    for (int n = 0; n < 4; ++n)
      acc[m][n] = (v8f){0.f, 0.f, 0.f, 0.f, 0.f, 0.f, 0.f, 0.f};

  // ---- prologue: stage K-step 0 into buffer 0 ----
  {
    uint4 a0 = *reinterpret_cast<const uint4*>(gA);
    uint4 a1 = *reinterpret_cast<const uint4*>(gA + 8);
    uint4 a2 = *reinterpret_cast<const uint4*>(gA + 16);
    uint4 a3 = *reinterpret_cast<const uint4*>(gA + 24);
    uint4 b0 = *reinterpret_cast<const uint4*>(gB);
    uint4 b1 = *reinterpret_cast<const uint4*>(gB + 8);
    *reinterpret_cast<uint4*>(sm + sAo)      = a0;
    *reinterpret_cast<uint4*>(sm + sAo + 8)  = a1;
    *reinterpret_cast<uint4*>(sm + sAo + 16) = a2;
    *reinterpret_cast<uint4*>(sm + sAo + 24) = a3;
    *reinterpret_cast<uint4*>(sm + sBo)      = b0;
    *reinterpret_cast<uint4*>(sm + sBo + 8)  = b1;
  }
  __syncthreads();

  int cur = 0;                                   // element offset of live buffer
#pragma unroll 1
  for (int kk = 0; kk < KSTEPS - 1; ++kk) {
    // prefetch next K-slab from global while computing on `cur`
    const __bf16* pa = gA + (size_t)(kk + 1) * 32;
    const __bf16* pb = gB + (size_t)(kk + 1) * 32;
    uint4 a0 = *reinterpret_cast<const uint4*>(pa);
    uint4 a1 = *reinterpret_cast<const uint4*>(pa + 8);
    uint4 a2 = *reinterpret_cast<const uint4*>(pa + 16);
    uint4 a3 = *reinterpret_cast<const uint4*>(pa + 24);
    uint4 b0 = *reinterpret_cast<const uint4*>(pb);
    uint4 b1 = *reinterpret_cast<const uint4*>(pb + 8);

    // ---- fragment loads (all first), then 16 back-to-back WMMAs ----
    {
      const __bf16* bufA = sm + cur + AOFF;
      const __bf16* bufB = sm + cur + BOFF;
      v16bf af[4], bv[4];
#pragma unroll
      for (int m = 0; m < 4; ++m) {
        const __bf16* p = bufA + (wm * 64 + m * 16 + lrow) * LDS_PITCH + lhalf * 8;
        v8bf lo = *reinterpret_cast<const v8bf*>(p);
        v8bf hi = *reinterpret_cast<const v8bf*>(p + 16);
        af[m] = __builtin_shufflevector(lo, hi, 0, 1, 2, 3, 4, 5, 6, 7,
                                        8, 9, 10, 11, 12, 13, 14, 15);
      }
#pragma unroll
      for (int n = 0; n < 4; ++n) {
        const __bf16* p = bufB + (wn * 64 + n * 16 + lrow) * LDS_PITCH + lhalf * 8;
        v8bf lo = *reinterpret_cast<const v8bf*>(p);
        v8bf hi = *reinterpret_cast<const v8bf*>(p + 16);
        bv[n] = __builtin_shufflevector(lo, hi, 0, 1, 2, 3, 4, 5, 6, 7,
                                        8, 9, 10, 11, 12, 13, 14, 15);
      }
#pragma unroll
      for (int m = 0; m < 4; ++m)
#pragma unroll
        for (int n = 0; n < 4; ++n)
          acc[m][n] = __builtin_amdgcn_wmma_f32_16x16x32_bf16(
              false, af[m], false, bv[n], (short)0, acc[m][n], false, false);
    }

    __syncthreads();
    const int nxt = cur ^ BUFSZ;
    *reinterpret_cast<uint4*>(sm + nxt + sAo)      = a0;
    *reinterpret_cast<uint4*>(sm + nxt + sAo + 8)  = a1;
    *reinterpret_cast<uint4*>(sm + nxt + sAo + 16) = a2;
    *reinterpret_cast<uint4*>(sm + nxt + sAo + 24) = a3;
    *reinterpret_cast<uint4*>(sm + nxt + sBo)      = b0;
    *reinterpret_cast<uint4*>(sm + nxt + sBo + 8)  = b1;
    __syncthreads();
    cur = nxt;
  }

  // ---- final K-step ----
  {
    const __bf16* bufA = sm + cur + AOFF;
    const __bf16* bufB = sm + cur + BOFF;
    v16bf af[4], bv[4];
#pragma unroll
    for (int m = 0; m < 4; ++m) {
      const __bf16* p = bufA + (wm * 64 + m * 16 + lrow) * LDS_PITCH + lhalf * 8;
      v8bf lo = *reinterpret_cast<const v8bf*>(p);
      v8bf hi = *reinterpret_cast<const v8bf*>(p + 16);
      af[m] = __builtin_shufflevector(lo, hi, 0, 1, 2, 3, 4, 5, 6, 7,
                                      8, 9, 10, 11, 12, 13, 14, 15);
    }
#pragma unroll
    for (int n = 0; n < 4; ++n) {
      const __bf16* p = bufB + (wn * 64 + n * 16 + lrow) * LDS_PITCH + lhalf * 8;
      v8bf lo = *reinterpret_cast<const v8bf*>(p);
      v8bf hi = *reinterpret_cast<const v8bf*>(p + 16);
      bv[n] = __builtin_shufflevector(lo, hi, 0, 1, 2, 3, 4, 5, 6, 7,
                                      8, 9, 10, 11, 12, 13, 14, 15);
    }
#pragma unroll
    for (int m = 0; m < 4; ++m)
#pragma unroll
      for (int n = 0; n < 4; ++n)
        acc[m][n] = __builtin_amdgcn_wmma_f32_16x16x32_bf16(
            false, af[m], false, bv[n], (short)0, acc[m][n], false, false);
  }

  // ---- epilogue: VGPR v -> row v + 8*lhalf, col = lane&15 ----
#pragma unroll
  for (int m = 0; m < 4; ++m) {
#pragma unroll
    for (int n = 0; n < 4; ++n) {
      const int r0 = bm * BM + wm * 64 + m * 16 + lhalf * 8;
      const int c0 = bn * BN + wn * 64 + n * 16 + lrow;
      float* p = C + (size_t)r0 * OUT_FEAT + c0;
#pragma unroll
      for (int v = 0; v < 8; ++v) p[(size_t)v * OUT_FEAT] = acc[m][n][v];
    }
  }
}

// ---------------------------------------------------------------------------
extern "C" void kernel_launch(void* const* d_in, const int* in_sizes, int n_in,
                              void* d_out, int out_size, void* d_ws, size_t ws_size,
                              hipStream_t stream) {
  const float* x  = (const float*)d_in[0];  // [8192,1024]
  const float* bw = (const float*)d_in[1];  // [1024,1024]
  const float* sw = (const float*)d_in[2];  // [1024,1024,8]
  const float* sc = (const float*)d_in[3];  // [1024,1024]
  const float* gr = (const float*)d_in[4];  // [1024,12]
  float* out = (float*)d_out;               // [8192,1024]

  __bf16* Abuf = (__bf16*)d_ws;                                        // 151 MB
  __bf16* Bbuf = (__bf16*)((char*)d_ws + (size_t)NROWS * KDIM * 2);    // +19 MB

  kan_prep_a<<<NROWS * 4, 256, 0, stream>>>(x, gr, Abuf);
  kan_prep_b<<<OUT_FEAT * 4, 256, 0, stream>>>(bw, sw, sc, Bbuf);

  dim3 gdim(OUT_FEAT / BN, NROWS / BM);     // (8, 32)
  kan_gemm<<<gdim, 256, 0, stream>>>(Abuf, Bbuf, out);
}
